// CoarseMatching_20280835572166
// MI455X (gfx1250) — matrix operand order repaired and checked by reading, not compiled, not run on hardware
//
#include <hip/hip_runtime.h>
#include <hip/hip_bf16.h>

typedef __attribute__((ext_vector_type(16))) __bf16 v16bf;
typedef __attribute__((ext_vector_type(8)))  __bf16 v8bf;
typedef __attribute__((ext_vector_type(8)))  float  v8f;

#define THR_F   0.01f
#define BORDER  2

__device__ __forceinline__ unsigned short f32_to_bf16_rne(float f) {
  unsigned u = __float_as_uint(f);
  u += 0x7FFFu + ((u >> 16) & 1u);      // round to nearest even
  return (unsigned short)(u >> 16);
}

// ---------------- 1) f32 -> bf16 conversion of both feature tensors ----------
__global__ __launch_bounds__(256) void k_convert(
    const float* __restrict__ f0, const float* __restrict__ f1,
    unsigned short* __restrict__ a, unsigned short* __restrict__ b, size_t n) {
  size_t i = (size_t)blockIdx.x * blockDim.x + threadIdx.x;
  if (i < n) {
    a[i] = f32_to_bf16_rne(f0[i]);
    b[i] = f32_to_bf16_rne(f1[i]);
  }
}

// ---------------- 2) sim = A * B^T * scale via bf16 WMMA ---------------------
// Block = 128 threads = 4 waves stacked in M. Wave tile = 48 (M) x 64 (N):
// 3 M-tiles x 4 N-tiles = 12 accumulators, K double-buffered & fully unrolled.
// Requires L % 192 == 0 and S % 64 == 0 (4800 = 25*192 = 75*64).
static constexpr int KD = 256;   // channel dim, fixed by the problem

__device__ __forceinline__ v16bf load_afrag(const unsigned short* __restrict__ arow,
                                            int k, int hi) {
  // A 16x32 bf16 layout: lanes 0-15 -> K = k + {0..7, 16..23};
  //                      lanes 16-31 -> K = k + {8..15, 24..31}
  union { v16bf v; v8bf h[2]; } u;
  u.h[0] = *(const v8bf*)(arow + k + hi * 8);
  u.h[1] = *(const v8bf*)(arow + k + 16 + hi * 8);
  return u.v;
}

__device__ __forceinline__ v16bf load_bfrag(const unsigned short* __restrict__ brow,
                                            int k, int hi) {
  // B 32x16 bf16 layout: lanes 0-15 -> K = k + {0..7, 16..23}; lanes 16-31 -> K = k + {8..15, 24..31}
  union { v16bf v; v8bf h[2]; } u;
  const unsigned short* p = brow + k + hi * 8;
  u.h[0] = *(const v8bf*)(p);
  u.h[1] = *(const v8bf*)(p + 16);
  return u.v;
}

__global__ __launch_bounds__(128) void k_simgemm(
    const unsigned short* __restrict__ A, const unsigned short* __restrict__ B,
    float* __restrict__ sim, int L, int S, float scale)
{
  const int lane = threadIdx.x & 31;
  const int wave = threadIdx.x >> 5;
  const int n16  = lane & 15;
  const int hi   = lane >> 4;
  const int row0 = blockIdx.x * 192 + wave * 48;
  const int col0 = blockIdx.y * 64;
  const int bat  = blockIdx.z;

  const unsigned short* Ab = A + (size_t)bat * L * KD;
  const unsigned short* Bb = B + (size_t)bat * S * KD;
  float* simb = sim + (size_t)bat * L * S;

  const unsigned short* arow[3];
  const unsigned short* brow[4];
#pragma unroll
  for (int i = 0; i < 3; ++i) arow[i] = Ab + (size_t)(row0 + i * 16 + n16) * KD;
#pragma unroll
  for (int t = 0; t < 4; ++t) brow[t] = Bb + (size_t)(col0 + t * 16 + n16) * KD;

  v8f acc[3][4] = {};

  v16bf af[3], bf[4], afn[3], bfn[4];
#pragma unroll
  for (int i = 0; i < 3; ++i) af[i] = load_afrag(arow[i], 0, hi);
#pragma unroll
  for (int t = 0; t < 4; ++t) bf[t] = load_bfrag(brow[t], 0, hi);

#pragma unroll
  for (int k = 0; k < KD; k += 32) {
    const bool more = (k + 32) < KD;
    if (more) {            // prefetch next K-step before consuming current one
#pragma unroll
      for (int i = 0; i < 3; ++i) afn[i] = load_afrag(arow[i], k + 32, hi);
#pragma unroll
      for (int t = 0; t < 4; ++t) bfn[t] = load_bfrag(brow[t], k + 32, hi);
    }
#pragma unroll
    for (int i = 0; i < 3; ++i)
#pragma unroll
      for (int t = 0; t < 4; ++t)
        acc[i][t] = __builtin_amdgcn_wmma_f32_16x16x32_bf16(
            false, af[i], false, bf[t], (short)0, acc[i][t], false, false);
    if (more) {
#pragma unroll
      for (int i = 0; i < 3; ++i) af[i] = afn[i];
#pragma unroll
      for (int t = 0; t < 4; ++t) bf[t] = bfn[t];
    }
  }

  // C/D layout: lane -> N = lane&15 ; VGPR v -> M = v + 8*(lane>>4)
#pragma unroll
  for (int i = 0; i < 3; ++i)
#pragma unroll
    for (int t = 0; t < 4; ++t)
#pragma unroll
      for (int v = 0; v < 8; ++v)
        simb[(size_t)(row0 + i * 16 + v + 8 * hi) * S + (col0 + t * 16 + n16)] =
            acc[i][t][v] * scale;
}

// ---------------- 3a) per-row online softmax stats (wave per row) ------------
__global__ __launch_bounds__(256) void k_rowstats(
    const float* __restrict__ sim, float* __restrict__ rmax,
    float* __restrict__ rsum, int S, int nrows)
{
  int g = blockIdx.x * 8 + (threadIdx.x >> 5);
  int lane = threadIdx.x & 31;
  if (g >= nrows) return;
  const float* row = sim + (size_t)g * S;
  float m = -3.4e38f, s = 0.f;
  for (int j = lane; j < S; j += 32) {
    float x = row[j];
    float nm = fmaxf(m, x);
    s = s * __expf(m - nm) + __expf(x - nm);
    m = nm;
  }
#pragma unroll
  for (int off = 16; off; off >>= 1) {
    float mo = __shfl_xor(m, off);
    float so = __shfl_xor(s, off);
    float nm = fmaxf(m, mo);
    s = s * __expf(m - nm) + so * __expf(mo - nm);
    m = nm;
  }
  if (lane == 0) { rmax[g] = m; rsum[g] = s; }
}

// ---------------- 3b) per-col stats (thread per column, coalesced) -----------
__global__ __launch_bounds__(256) void k_colstats(
    const float* __restrict__ sim, float* __restrict__ cmax, float* __restrict__ csum,
    unsigned char* __restrict__ cvalid,
    const int* __restrict__ h1c, const int* __restrict__ w1c, int L, int S)
{
  int s = blockIdx.x * blockDim.x + threadIdx.x;
  int b = blockIdx.y;
  if (s >= S) return;
  const float* base = sim + (size_t)b * L * S + s;
  float m = -3.4e38f, sm = 0.f;
  for (int l = 0; l < L; ++l) {
    float x = base[(size_t)l * S];
    float nm = fmaxf(m, x);
    sm = sm * __expf(m - nm) + __expf(x - nm);
    m = nm;
  }
  cmax[b * S + s] = m;
  csum[b * S + s] = sm;
  if (b == 0) {
    int w = *w1c, h = *h1c;
    int i1 = s / w, j1 = s % w;
    cvalid[s] = (i1 >= BORDER && i1 < h - BORDER &&
                 j1 >= BORDER && j1 < w - BORDER) ? 1 : 0;
  }
}

// ---------------- 4) conf = exp(2*sim - rm - cm)/(rs*cs), + row conf-max -----
__global__ __launch_bounds__(256) void k_conf(
    float* __restrict__ simconf, const float* __restrict__ rmax,
    const float* __restrict__ rsum, const float* __restrict__ cmax,
    const float* __restrict__ csum, float* __restrict__ rcm,
    int L, int S, int nrows)
{
  int g = blockIdx.x * 8 + (threadIdx.x >> 5);
  int lane = threadIdx.x & 31;
  if (g >= nrows) return;
  int n = g / L;
  float* row = simconf + (size_t)g * S;
  const float* cm = cmax + (size_t)n * S;
  const float* cs = csum + (size_t)n * S;
  float rm = rmax[g];
  float ri = 1.0f / rsum[g];
  float mx = 0.f;
  for (int j = lane; j < S; j += 32) {
    float c = __expf(2.f * row[j] - rm - cm[j]) * (ri / cs[j]);
    row[j] = c;
    mx = fmaxf(mx, c);
  }
#pragma unroll
  for (int off = 16; off; off >>= 1) mx = fmaxf(mx, __shfl_xor(mx, off));
  if (lane == 0) rcm[g] = mx;
}

// ---------------- 5) per-column conf max -------------------------------------
__global__ __launch_bounds__(256) void k_colconfmax(
    const float* __restrict__ conf, float* __restrict__ ccm, int L, int S)
{
  int s = blockIdx.x * blockDim.x + threadIdx.x;
  int b = blockIdx.y;
  if (s >= S) return;
  const float* base = conf + (size_t)b * L * S + s;
  float m = 0.f;
  for (int l = 0; l < L; ++l) m = fmaxf(m, base[(size_t)l * S]);
  ccm[b * S + s] = m;
}

// ---------------- 6) match extraction (wave per row) -------------------------
__global__ __launch_bounds__(256) void k_match(
    const float* __restrict__ conf, const float* __restrict__ rcm,
    const float* __restrict__ ccm, const unsigned char* __restrict__ cvalid,
    const int* __restrict__ h0c, const int* __restrict__ w0c,
    float* __restrict__ mask_v, int* __restrict__ all_j, float* __restrict__ mconf,
    int L, int S, int nrows)
{
  int g = blockIdx.x * 8 + (threadIdx.x >> 5);
  int lane = threadIdx.x & 31;
  if (g >= nrows) return;
  int n = g / L, l = g - n * L;
  int w0 = *w0c, h0 = *h0c;
  int i0 = l / w0, j0 = l - i0 * w0;
  bool rowok = (i0 >= BORDER && i0 < h0 - BORDER &&
                j0 >= BORDER && j0 < w0 - BORDER);
  const float* row = conf + (size_t)g * S;
  const float* cc = ccm + (size_t)n * S;
  float rm = rcm[g];
  int best = 0x7FFFFFFF;
  if (rowok) {
    for (int j = lane; j < S; j += 32) {
      float c = row[j];
      if ((c > THR_F) && cvalid[j] && (c == rm) && (c == cc[j]) && (j < best))
        best = j;
    }
  }
#pragma unroll
  for (int off = 16; off; off >>= 1) best = min(best, __shfl_xor(best, off));
  if (lane == 0) {
    bool found = best < S;
    mask_v[g] = found ? 1.f : 0.f;     // argmax of all-False bool == 0 in reference
    all_j[g]  = found ? best : 0;
    mconf[g]  = found ? row[best] : 0.f;
  }
}

// =============================================================================
extern "C" void kernel_launch(void* const* d_in, const int* in_sizes, int n_in,
                              void* d_out, int out_size, void* d_ws, size_t ws_size,
                              hipStream_t stream) {
  const int N = 2, C = 256;
  const int L = in_sizes[0] / (N * C);          // 4800
  const int S = in_sizes[1] / (N * C);          // 4800
  const float scale = 1.0f / ((float)C * 0.1f); // 1/(C*TEMP)

  const float* f0 = (const float*)d_in[0];
  const float* f1 = (const float*)d_in[1];
  const int* h0c = (const int*)d_in[2];
  const int* w0c = (const int*)d_in[3];
  const int* h1c = (const int*)d_in[4];
  const int* w1c = (const int*)d_in[5];

  // workspace layout
  unsigned short* Abf = (unsigned short*)d_ws;                 // N*L*C bf16
  unsigned short* Bbf = Abf + (size_t)N * L * C;               // N*S*C bf16
  float* rmax = (float*)(Bbf + (size_t)N * S * C);
  float* rsum = rmax + (size_t)N * L;
  float* cmax = rsum + (size_t)N * L;
  float* csum = cmax + (size_t)N * S;
  float* rcm  = csum + (size_t)N * S;
  float* ccm  = rcm + (size_t)N * L;
  unsigned char* cval = (unsigned char*)(ccm + (size_t)N * S);

  // output layout: conf [N*L*S] f32, mask_v [N*L], all_j [N*L] i32, mconf [N*L]
  float* conf   = (float*)d_out;
  float* mask_v = conf + (size_t)N * L * S;
  int*   all_j  = (int*)(mask_v + (size_t)N * L);
  float* mconf  = (float*)(all_j + (size_t)N * L);

  const size_t nf = (size_t)N * L * C;
  const int nrows = N * L;

  k_convert<<<(unsigned)((nf + 255) / 256), 256, 0, stream>>>(f0, f1, Abf, Bbf, nf);

  dim3 gg(L / 192, S / 64, N);
  k_simgemm<<<gg, 128, 0, stream>>>(Abf, Bbf, conf, L, S, scale);

  k_rowstats<<<(nrows + 7) / 8, 256, 0, stream>>>(conf, rmax, rsum, S, nrows);

  dim3 gc((S + 255) / 256, N);
  k_colstats<<<gc, 256, 0, stream>>>(conf, cmax, csum, cval, h1c, w1c, L, S);

  k_conf<<<(nrows + 7) / 8, 256, 0, stream>>>(conf, rmax, rsum, cmax, csum, rcm,
                                              L, S, nrows);

  k_colconfmax<<<gc, 256, 0, stream>>>(conf, ccm, L, S);

  k_match<<<(nrows + 7) / 8, 256, 0, stream>>>(conf, rcm, ccm, cval, h0c, w0c,
                                               mask_v, all_j, mconf, L, S, nrows);
}
